// MultiHeadAttention_16647293240130
// MI455X (gfx1250) — matrix operand order repaired and checked by reading, not compiled, not run on hardware
//
#include <hip/hip_runtime.h>

// ---------------------------------------------------------------------------
// MHA forward for MI455X (gfx1250): bf16 WMMA + async global->LDS staging.
// D=1024, H=16, hd=64, S=2048, B=2.
// ---------------------------------------------------------------------------

typedef __bf16 bf16;
typedef bf16  v16bf __attribute__((ext_vector_type(16)));
typedef bf16  v8bf  __attribute__((ext_vector_type(8)));
typedef float v8f   __attribute__((ext_vector_type(8)));
typedef int   v4i   __attribute__((ext_vector_type(4)));

#define DCONST 1024
#define HCONST 16
#define HD     64
#define SCONST 2048
#define BCONST 2
#define KDIM   1024
#define BS     (BCONST * SCONST)   // 4096 rows for all projection GEMMs

#define KSTRIDE 72   // LDS row stride (elements) for 64-wide tiles (bank spread)
#define BSTRIDE 40   // LDS row stride (elements) for 32-wide tiles

#if defined(__has_builtin)
#  if __has_builtin(__builtin_amdgcn_global_load_async_to_lds_b128)
#    define HAVE_ASYNC_BUILTIN 1
#  endif
#  if __has_builtin(__builtin_amdgcn_s_wait_asynccnt)
#    define HAVE_WAIT_ASYNC_BUILTIN 1
#  endif
#endif

static __device__ __forceinline__ void async_cp16(const bf16* g, bf16* l) {
#if defined(HAVE_ASYNC_BUILTIN)
  __builtin_amdgcn_global_load_async_to_lds_b128(
      (__attribute__((address_space(1))) v4i*)g,
      (__attribute__((address_space(3))) v4i*)l, 0, 0);
#else
  // VDST = LDS byte offset (low 32 bits of the flat LDS pointer), VADDR = 64b
  unsigned loff = (unsigned)(size_t)l;
  asm volatile("global_load_async_to_lds_b128 %0, %1, off"
               :: "v"(loff), "v"(g) : "memory");
#endif
}

template <int N>
static __device__ __forceinline__ void wait_async() {
#if defined(HAVE_WAIT_ASYNC_BUILTIN)
  __builtin_amdgcn_s_wait_asynccnt(N);
#else
  asm volatile("s_wait_asynccnt %0" :: "i"(N) : "memory");
#endif
}

static __device__ __forceinline__ v16bf cat8(v8bf lo, v8bf hi) {
  return __builtin_shufflevector(lo, hi, 0, 1, 2, 3, 4, 5, 6, 7,
                                 8, 9, 10, 11, 12, 13, 14, 15);
}

static __device__ __forceinline__ v8f wmma_bf16(v16bf a, v16bf b, v8f c) {
  // D = A(16x32) * B(32x16) + C, fp32 accumulate
  return __builtin_amdgcn_wmma_f32_16x16x32_bf16(
      /*neg_a=*/false, a, /*neg_b=*/false, b,
      /*c_mod=*/(short)0, c, /*reuse_a=*/false, /*reuse_b=*/false);
}

// A fragment (16x32 bf16): lane = M (row), per-lane two 16B loads.
// K(e) = (e/8)*16 + half*8 + e%8  ->  lo @ k0+half*8, hi @ k0+16+half*8
static __device__ __forceinline__ v16bf load_afrag(const bf16* row, int k0,
                                                   int half) {
  v8bf lo = *(const v8bf*)(row + k0 + half * 8);
  v8bf hi = *(const v8bf*)(row + k0 + 16 + half * 8);
  return cat8(lo, hi);
}

// B fragment (32x16 bf16): lane = N (col), K(e) = (e/8)*16 + half*8 + e%8
static __device__ __forceinline__ v16bf load_bfrag(const bf16* colrow, int k0,
                                                   int half) {
  v8bf lo = *(const v8bf*)(colrow + k0 + half * 8);
  v8bf hi = *(const v8bf*)(colrow + k0 + 16 + half * 8);
  return cat8(lo, hi);
}

// ---------------------------------------------------------------------------
// Pack kernels
// ---------------------------------------------------------------------------

// (S,B,D) fp32 -> (B,S,D) bf16, 8 elements / thread
__global__ void mha_transpose_cast(const float* __restrict__ src,
                                   bf16* __restrict__ dst) {
  int i    = blockIdx.x * blockDim.x + threadIdx.x;
  int dv   = i % (DCONST / 8);
  int rest = i / (DCONST / 8);
  int b    = rest % BCONST;
  int s    = rest / BCONST;
  const float4* sp =
      (const float4*)(src + ((size_t)s * BCONST + b) * DCONST + dv * 8);
  float4 f0 = sp[0];
  float4 f1 = sp[1];
  v8bf o;
  o[0] = (bf16)f0.x; o[1] = (bf16)f0.y; o[2] = (bf16)f0.z; o[3] = (bf16)f0.w;
  o[4] = (bf16)f1.x; o[5] = (bf16)f1.y; o[6] = (bf16)f1.z; o[7] = (bf16)f1.w;
  *(v8bf*)(dst + ((size_t)b * SCONST + s) * DCONST + dv * 8) = o;
}

// fp32 -> bf16 elementwise (weights), 8 elements / thread
__global__ void mha_cast(const float* __restrict__ src, bf16* __restrict__ dst) {
  int i = blockIdx.x * blockDim.x + threadIdx.x;
  const float4* sp = (const float4*)(src + (size_t)i * 8);
  float4 f0 = sp[0];
  float4 f1 = sp[1];
  v8bf o;
  o[0] = (bf16)f0.x; o[1] = (bf16)f0.y; o[2] = (bf16)f0.z; o[3] = (bf16)f0.w;
  o[4] = (bf16)f1.x; o[5] = (bf16)f1.y; o[6] = (bf16)f1.z; o[7] = (bf16)f1.w;
  *(v8bf*)(dst + (size_t)i * 8) = o;
}

// ---------------------------------------------------------------------------
// GEMM: O = relu(X @ W^T + bias); X:(BS,K) bf16 rowmajor, W:(N,K) bf16 rowmajor.
// Block = 8 waves: tile 256(M) x 64(N). Shared 64x32 B tile async-staged in
// LDS, double-buffered; A fragments stream from global per wave.
// mode 0: out_bf[(b,h,s,d')] ; mode 1: out_bf[(b,h,d',s)] ; mode 2: fp32 (s,b,n)
// ---------------------------------------------------------------------------
__global__ __launch_bounds__(256) void mha_gemm_bias_relu(
    const bf16* __restrict__ X, const bf16* __restrict__ W,
    const float* __restrict__ bias, bf16* __restrict__ outb,
    float* __restrict__ outf, int mode) {
  __shared__ __align__(16) bf16 bsh[2][64 * BSTRIDE];

  const int tid  = threadIdx.x;
  const int wave = tid >> 5;
  const int lane = tid & 31;
  const int lc   = lane & 15;
  const int half = lane >> 4;
  const int m0   = (blockIdx.x * 8 + wave) * 32;
  const int n0   = blockIdx.y * 64;

  // each thread owns one 16B chunk of the 64x32 B tile: row = tid/4, sub = tid%4
  const int brow = tid >> 2;
  const int bsub = tid & 3;
  const bf16* bsrc = W + (size_t)(n0 + brow) * KDIM + bsub * 8;
  bf16*       bdst0 = &bsh[0][brow * BSTRIDE + bsub * 8];
  bf16*       bdst1 = &bsh[1][brow * BSTRIDE + bsub * 8];

  v8f acc[2][4] = {};

  async_cp16(bsrc, bdst0);  // k0 = 0 -> buffer 0

  for (int k0 = 0; k0 < KDIM; k0 += 32) {
    const int cur = (k0 >> 5) & 1;
    if (k0 + 32 < KDIM) {
      async_cp16(bsrc + (k0 + 32), cur ? bdst0 : bdst1);
      wait_async<1>();
    } else {
      wait_async<0>();
    }
    __syncthreads();

    v16bf aA[2];
#pragma unroll
    for (int mi = 0; mi < 2; ++mi) {
      const bf16* ap = X + (size_t)(m0 + mi * 16 + lc) * KDIM;
      aA[mi] = load_afrag(ap, k0, half);
    }
#pragma unroll
    for (int t = 0; t < 4; ++t) {
      const bf16* wp = &bsh[cur][(t * 16 + lc) * BSTRIDE];
      v16bf bB = load_bfrag(wp, 0, half);
      acc[0][t] = wmma_bf16(aA[0], bB, acc[0][t]);
      acc[1][t] = wmma_bf16(aA[1], bB, acc[1][t]);
    }
    __syncthreads();  // protect buffer reuse across waves
  }

#pragma unroll
  for (int t = 0; t < 4; ++t) {
    const int n  = n0 + t * 16 + lc;
    const float bv = bias[n];
#pragma unroll
    for (int mi = 0; mi < 2; ++mi) {
#pragma unroll
      for (int r = 0; r < 8; ++r) {
        const int m = m0 + mi * 16 + r + 8 * half;
        float v = acc[mi][t][r] + bv;
        v = v > 0.0f ? v : 0.0f;
        const int b = m / SCONST;
        const int s = m % SCONST;
        if (mode == 0) {
          const int hh = n >> 6, dd = n & 63;
          outb[(((size_t)b * HCONST + hh) * SCONST + s) * HD + dd] = (bf16)v;
        } else if (mode == 1) {
          const int hh = n >> 6, dd = n & 63;
          outb[(((size_t)b * HCONST + hh) * HD + dd) * SCONST + s] = (bf16)v;
        } else {
          outf[((size_t)s * BCONST + b) * DCONST + n] = v;
        }
      }
    }
  }
}

// ---------------------------------------------------------------------------
// Flash attention. Block = 8 waves, all on the same (b,h): wave w owns query
// rows [m0, m0+16). 64-key K/V tiles are shared by the whole block: async
// global->LDS, double-buffered. P relayout (C->A) via per-wave LDS staging.
// qh,kh:(B,H,S,64) ; vT:(B,H,64,S) ; y:(B,S,D) all bf16.
// ---------------------------------------------------------------------------
__global__ __launch_bounds__(256) void mha_flash_attn(
    const bf16* __restrict__ qh, const bf16* __restrict__ kh,
    const bf16* __restrict__ vT, bf16* __restrict__ y) {
  __shared__ __align__(16) bf16 ks[2][64 * KSTRIDE];   // 18 KB
  __shared__ __align__(16) bf16 vs[2][64 * KSTRIDE];   // 18 KB
  __shared__ __align__(16) bf16 psh[8][16 * KSTRIDE];  // 18 KB

  const int tid  = threadIdx.x;
  const int wave = tid >> 5;
  const int lane = tid & 31;
  const int lc   = lane & 15;
  const int half = lane >> 4;

  const int g    = blockIdx.x;       // 0..511
  const int mgrp = g & 15;
  const int bh   = g >> 4;           // 0..31
  const int h    = bh & 15;
  const int b    = bh >> 4;
  const int m0   = (mgrp * 8 + wave) * 16;

  const bf16* qbase = qh + (((size_t)b * HCONST + h) * SCONST + m0) * HD;
  const bf16* kbase = kh + (((size_t)b * HCONST + h) * SCONST) * HD;
  const bf16* vbase = vT + (((size_t)b * HCONST + h) * HD) * SCONST;

  // Q fragments (K = 64 -> two 16x32 A-frags), loaded once
  v16bf qA[2];
#pragma unroll
  for (int kk = 0; kk < 2; ++kk)
    qA[kk] = load_afrag(qbase + (size_t)lc * HD, kk * 32, half);

  float mrow[8], lrow[8];
  v8f oacc[4] = {};
#pragma unroll
  for (int r = 0; r < 8; ++r) { mrow[r] = -1e30f; lrow[r] = 0.0f; }

  // cooperative async tile copy: per thread 2 K-chunks + 2 V-chunks (16B each)
  auto issue_tiles = [&](int kb, int buf) {
#pragma unroll
    for (int j = 0; j < 2; ++j) {
      const int cc  = tid + j * 256;     // 0..511
      const int row = cc >> 3;
      const int sub = cc & 7;
      async_cp16(kbase + (size_t)(kb + row) * HD + sub * 8,
                 &ks[buf][row * KSTRIDE + sub * 8]);
    }
#pragma unroll
    for (int j = 0; j < 2; ++j) {
      const int cc  = tid + j * 256;
      const int row = cc >> 3;           // d' 0..63
      const int sub = cc & 7;
      async_cp16(vbase + (size_t)row * SCONST + kb + sub * 8,
                 &vs[buf][row * KSTRIDE + sub * 8]);
    }
  };

  issue_tiles(0, 0);

  for (int kb = 0; kb < SCONST; kb += 64) {
    const int cur = (kb >> 6) & 1;
    if (kb + 64 < SCONST) {
      issue_tiles(kb + 64, cur ^ 1);
      wait_async<4>();   // async loads retire in order: current 4 are done
    } else {
      wait_async<0>();
    }
    __syncthreads();

    // ---- S = (Q @ K^T) * (1/sqrt(64)), 16x64 tile as 4 C-frags ----
    v8f sfrag[4];
#pragma unroll
    for (int t = 0; t < 4; ++t) {
      const bf16* kp = &ks[cur][(t * 16 + lc) * KSTRIDE];
      v8f acc = {};
#pragma unroll
      for (int kk = 0; kk < 2; ++kk) {
        v16bf bB = load_bfrag(kp, kk * 32, half);
        acc = wmma_bf16(qA[kk], bB, acc);
      }
#pragma unroll
      for (int r = 0; r < 8; ++r) acc[r] *= 0.125f;
      sfrag[t] = acc;
    }

    // ---- online softmax: row max across 16 lanes (per half), rescale ----
#pragma unroll
    for (int r = 0; r < 8; ++r) {
      float v = fmaxf(fmaxf(sfrag[0][r], sfrag[1][r]),
                      fmaxf(sfrag[2][r], sfrag[3][r]));
      v = fmaxf(v, __shfl_xor(v, 1, 32));
      v = fmaxf(v, __shfl_xor(v, 2, 32));
      v = fmaxf(v, __shfl_xor(v, 4, 32));
      v = fmaxf(v, __shfl_xor(v, 8, 32));
      const float mnew  = fmaxf(mrow[r], v);
      const float alpha = __expf(mrow[r] - mnew);
      mrow[r] = mnew;
      lrow[r] *= alpha;
#pragma unroll
      for (int t = 0; t < 4; ++t) oacc[t][r] *= alpha;
    }

    // ---- P = exp(S - m), row-sum into l, stage P (bf16) in LDS ----
#pragma unroll
    for (int t = 0; t < 4; ++t) {
#pragma unroll
      for (int r = 0; r < 8; ++r) {
        const float p = __expf(sfrag[t][r] - mrow[r]);
        float s = p;
        s += __shfl_xor(s, 1, 32);
        s += __shfl_xor(s, 2, 32);
        s += __shfl_xor(s, 4, 32);
        s += __shfl_xor(s, 8, 32);
        lrow[r] += s;
        psh[wave][(r + 8 * half) * KSTRIDE + t * 16 + lc] = (bf16)p;
      }
    }

    // ---- reload P in A-frag layout (C->A relayout through LDS) ----
    v16bf pA[2];
#pragma unroll
    for (int kk = 0; kk < 2; ++kk)
      pA[kk] = load_afrag(&psh[wave][lc * KSTRIDE], kk * 32, half);

    // ---- O += P @ V ----
#pragma unroll
    for (int t = 0; t < 4; ++t) {
      const bf16* vp = &vs[cur][(t * 16 + lc) * KSTRIDE];
#pragma unroll
      for (int kk = 0; kk < 2; ++kk) {
        v16bf vB = load_bfrag(vp, kk * 32, half);
        oacc[t] = wmma_bf16(pA[kk], vB, oacc[t]);
      }
    }
    __syncthreads();  // all waves done with ks/vs[cur] before it is rewritten
  }

  // ---- normalize and store y (B,S,D) bf16 ----
#pragma unroll
  for (int r = 0; r < 8; ++r) {
    const float inv = 1.0f / lrow[r];
    const int m = m0 + r + 8 * half;
#pragma unroll
    for (int t = 0; t < 4; ++t) {
      y[((size_t)b * SCONST + m) * DCONST + h * 64 + t * 16 + lc] =
          (bf16)(oacc[t][r] * inv);
    }
  }
}

// ---------------------------------------------------------------------------
// Host launch
// ---------------------------------------------------------------------------
extern "C" void kernel_launch(void* const* d_in, const int* in_sizes, int n_in,
                              void* d_out, int out_size, void* d_ws, size_t ws_size,
                              hipStream_t stream) {
  const float* q  = (const float*)d_in[0];
  const float* k  = (const float*)d_in[1];
  const float* v  = (const float*)d_in[2];
  const float* Wq = (const float*)d_in[3];
  const float* Wk = (const float*)d_in[4];
  const float* Wv = (const float*)d_in[5];
  const float* Wo = (const float*)d_in[6];
  const float* bq = (const float*)d_in[7];
  const float* bk = (const float*)d_in[8];
  const float* bv = (const float*)d_in[9];
  const float* bo = (const float*)d_in[10];
  float* out = (float*)d_out;

  const size_t BSD = (size_t)BCONST * SCONST * DCONST;  // 4,194,304
  const size_t DD  = (size_t)DCONST * DCONST;           // 1,048,576

  bf16* xq  = (bf16*)d_ws;
  bf16* xk  = xq + BSD;
  bf16* xv  = xk + BSD;
  bf16* wqb = xv + BSD;
  bf16* wkb = wqb + DD;
  bf16* wvb = wkb + DD;
  bf16* wob = wvb + DD;
  bf16* qhp = wob + DD;   // (B,H,S,64)
  bf16* khp = qhp + BSD;  // (B,H,S,64)
  bf16* vTp = khp + BSD;  // (B,H,64,S)
  bf16* yb  = vTp + BSD;  // (B,S,D)

  dim3 blk(256);

  // 1) pack inputs
  const int tpack = (int)(BSD / 8);  // 524288 threads
  mha_transpose_cast<<<tpack / 256, blk, 0, stream>>>(q, xq);
  mha_transpose_cast<<<tpack / 256, blk, 0, stream>>>(k, xk);
  mha_transpose_cast<<<tpack / 256, blk, 0, stream>>>(v, xv);
  const int twcast = (int)(DD / 8);  // 131072 threads
  mha_cast<<<twcast / 256, blk, 0, stream>>>(Wq, wqb);
  mha_cast<<<twcast / 256, blk, 0, stream>>>(Wk, wkb);
  mha_cast<<<twcast / 256, blk, 0, stream>>>(Wv, wvb);
  mha_cast<<<twcast / 256, blk, 0, stream>>>(Wo, wob);

  // 2) Q/K/V projections (bias + ReLU fused), head layouts
  dim3 ggrid(BS / 256, DCONST / 64);  // (16,16)
  mha_gemm_bias_relu<<<ggrid, blk, 0, stream>>>(xq, wqb, bq, qhp, nullptr, 0);
  mha_gemm_bias_relu<<<ggrid, blk, 0, stream>>>(xk, wkb, bk, khp, nullptr, 0);
  mha_gemm_bias_relu<<<ggrid, blk, 0, stream>>>(xv, wvb, bv, vTp, nullptr, 1);

  // 3) flash attention: 512 blocks x 8 waves
  mha_flash_attn<<<512, blk, 0, stream>>>(qhp, khp, vTp, yb);

  // 4) output projection -> fp32 (S,B,D)
  mha_gemm_bias_relu<<<ggrid, blk, 0, stream>>>(yb, wob, bo, nullptr, out, 2);
}
